// CrossModalAttention_51539607552817
// MI455X (gfx1250) — matrix operand order repaired and checked by reading, not compiled, not run on hardware
//
#include <hip/hip_runtime.h>

// CrossModalAttention for MI455X (gfx1250, wave32, WMMA bf16 path).
// Pipeline: f32->bf16 convert (+weight transpose), QKV GEMM (WMMA),
// flash attention with transposed-score trick (WMMA), output GEMM (WMMA),
// residual+LayerNorm. Straight-copy LDS staging uses CDNA5 async
// global->LDS DMA (ASYNCcnt). Needs ~50 MiB workspace.

#define NB 4
#define NS 2048
#define NE 512
#define NH 8
#define NHD 64
#define NM (NB * NS) // 8192

typedef unsigned short u16;
typedef __attribute__((ext_vector_type(4)))  unsigned short u16x4;
typedef __attribute__((ext_vector_type(8)))  unsigned short u16x8;
typedef __attribute__((ext_vector_type(16))) __bf16 bf16x16;
typedef __attribute__((ext_vector_type(8)))  float f32x8;

union Frag { bf16x16 v; u16 u[16]; u16x8 h[2]; unsigned w[8]; };

__device__ __forceinline__ u16 f2bf(float f) {
  union { float f; unsigned u; } x; x.f = f;
  unsigned r = x.u + 0x7FFFu + ((x.u >> 16) & 1u); // round-to-nearest-even
  return (u16)(r >> 16);
}

__device__ __forceinline__ unsigned f2bf_pk(float lo, float hi) {
  return (unsigned)f2bf(lo) | ((unsigned)f2bf(hi) << 16);
}

__device__ __forceinline__ f32x8 bfwmma(Frag a, Frag b, f32x8 c) {
  // (neg_a, A, neg_b, B, c_mod, C, reuse_a, reuse_b)
  return __builtin_amdgcn_wmma_f32_16x16x32_bf16(false, a.v, false, b.v,
                                                 (short)0, c, false, false);
}

// Async DMA: 16 bytes global -> LDS per lane, tracked by ASYNCcnt.
// Generic pointer to __shared__ carries the LDS offset in its low 32 bits
// (ISA 10.2: LDS aperture address = addr[31:0]).
__device__ __forceinline__ void async_copy_b128(void* lds, const void* gaddr) {
  unsigned l = (unsigned)(unsigned long long)lds;
  asm volatile("global_load_async_to_lds_b128 %0, %1, off"
               :: "v"(l), "v"(gaddr) : "memory");
}
__device__ __forceinline__ void async_wait0() {
  asm volatile("s_wait_asynccnt 0x0" ::: "memory");
}

// ---------------------------------------------------------------- convert
__global__ __launch_bounds__(256)
void cvt_kernel(const float* __restrict__ in, u16* __restrict__ out, int n) {
  int i = (blockIdx.x * 256 + threadIdx.x) * 4;
  if (i + 3 < n) {
    float4 f = *(const float4*)(in + i);
    u16x4 o = { f2bf(f.x), f2bf(f.y), f2bf(f.z), f2bf(f.w) };
    *(u16x4*)(out + i) = o;
  }
}

// 512x512 f32 -> bf16 transpose: out[n*NE + k] = bf16(in[k*NE + n])
__global__ __launch_bounds__(256)
void cvt_t_kernel(const float* __restrict__ in, u16* __restrict__ out) {
  __shared__ float tile[32][33];
  int bx = blockIdx.x * 32; // k base
  int by = blockIdx.y * 32; // n base
  int tx = threadIdx.x & 31, ty = threadIdx.x >> 5; // 32x8
#pragma unroll
  for (int r = 0; r < 32; r += 8)
    tile[ty + r][tx] = in[(size_t)(bx + ty + r) * NE + by + tx];
  __syncthreads();
#pragma unroll
  for (int r = 0; r < 32; r += 8)
    out[(size_t)(by + ty + r) * NE + bx + tx] = f2bf(tile[tx][ty + r]);
}

// ---------------------------------------------------------- QKV projection
// C = X(bf16)[8192x512] @ W + bias, W given TRANSPOSED (Wt[n][k], bf16).
// Q additionally pre-scaled by modal_compat[h]/sqrt(HD) (folded attention
// score scale). Scatter to [B,H,S,HD] bf16.
// Block tile 128x64, 8 waves, wave 32x32 (2x2 WMMA tiles), K step 32.
__global__ __launch_bounds__(256)
void qkv_gemm_kernel(const u16* __restrict__ Xq, const u16* __restrict__ Xk,
                     const u16* __restrict__ Xv,
                     const u16* __restrict__ WqT, const u16* __restrict__ WkT,
                     const u16* __restrict__ WvT,
                     const float* __restrict__ bq, const float* __restrict__ bk,
                     const float* __restrict__ bv,
                     const float* __restrict__ compat,
                     u16* __restrict__ Qp, u16* __restrict__ Kp,
                     u16* __restrict__ Vp) {
  const int z = blockIdx.z;
  const u16* X = (z == 0) ? Xq : (z == 1) ? Xk : Xv;
  const u16* Wt = (z == 0) ? WqT : (z == 1) ? WkT : WvT;
  const float* bias = (z == 0) ? bq : (z == 1) ? bk : bv;
  u16* Out = (z == 0) ? Qp : (z == 1) ? Kp : Vp;

  __shared__ u16 As[128][32];  // 8 KiB, row-major (K contiguous)
  __shared__ u16 BsT[64][32];  // 4 KiB, [n][k] (K contiguous)

  const int tid = threadIdx.x;
  const int lane = tid & 31, wave = tid >> 5;
  const int wm = wave >> 1, wn = wave & 1;
  const int ln15 = lane & 15, half = lane >> 4;
  const int blockM = blockIdx.x * 128;
  const int blockN = blockIdx.y * 64;

  f32x8 acc[2][2] = {};

  for (int k0 = 0; k0 < NE; k0 += 32) {
    __syncthreads();
#pragma unroll
    for (int p = 0; p < 2; ++p) { // A tile: 128 rows x 32 k (async DMA)
      int c = tid + p * 256;
      int r = c >> 2, off = (c & 3) * 8;
      async_copy_b128(&As[r][off], &X[(size_t)(blockM + r) * NE + k0 + off]);
    }
    { // B tile: 64 n x 32 k (async DMA from transposed weights)
      int r = tid >> 2, off = (tid & 3) * 8;
      async_copy_b128(&BsT[r][off], &Wt[(size_t)(blockN + r) * NE + k0 + off]);
      if (k0 + 32 < NE) {
        __builtin_prefetch(&Wt[(size_t)(blockN + r) * NE + k0 + 32 + off], 0, 3);
        __builtin_prefetch(&X[(size_t)(blockM + (tid >> 2)) * NE + k0 + 32], 0, 3);
      }
    }
    async_wait0();
    __syncthreads();

    Frag a[2], b[2];
#pragma unroll
    for (int t = 0; t < 2; ++t) {
      int rA = wm * 32 + t * 16 + ln15;
      a[t].h[0] = *(const u16x8*)&As[rA][half * 8];
      a[t].h[1] = *(const u16x8*)&As[rA][16 + half * 8];
      int cB = wn * 32 + t * 16 + ln15;
      b[t].h[0] = *(const u16x8*)&BsT[cB][half * 8];
      b[t].h[1] = *(const u16x8*)&BsT[cB][16 + half * 8];
    }
#pragma unroll
    for (int tm = 0; tm < 2; ++tm)
#pragma unroll
      for (int tn = 0; tn < 2; ++tn)
        acc[tm][tn] = bfwmma(a[tm], b[tn], acc[tm][tn]);
  }

#pragma unroll
  for (int tm = 0; tm < 2; ++tm) {
#pragma unroll
    for (int tn = 0; tn < 2; ++tn) {
      int gn = blockN + wn * 32 + tn * 16 + ln15;
      float bias_v = bias[gn];
      int hh = gn >> 6, dd = gn & 63;
      float scl = (z == 0) ? 0.125f * compat[hh] : 1.0f; // 1/sqrt(64)*compat
#pragma unroll
      for (int j = 0; j < 8; ++j) {
        int gm = blockM + wm * 32 + tm * 16 + j + 8 * half;
        int bb = gm >> 11, ss = gm & 2047;
        float vv = (acc[tm][tn][j] + bias_v) * scl;
        Out[(((size_t)bb * NH + hh) * NS + ss) * NHD + dd] = f2bf(vv);
      }
    }
  }
}

// --------------------------------------------------------- flash attention
// grid: (S/128, B*H). 8 waves/block, wave owns 16 query rows.
// Scores are computed TRANSPOSED (T = K_chunk @ Q^T) so the exp'd
// probability tile is already in WMMA A-fragment layout in-lane: no LDS
// round-trip, no cross-lane data movement for P.
__global__ __launch_bounds__(256)
void attn_kernel(const u16* __restrict__ Qp, const u16* __restrict__ Kp,
                 const u16* __restrict__ Vp, u16* __restrict__ Att) {
  const int bh = blockIdx.y;
  const int hh = bh & (NH - 1), bb = bh >> 3;
  const u16* Q = Qp + (size_t)bh * NS * NHD;
  const u16* K = Kp + (size_t)bh * NS * NHD;
  const u16* V = Vp + (size_t)bh * NS * NHD;

  const int tid = threadIdx.x, lane = tid & 31, wave = tid >> 5;
  const int ln15 = lane & 15, half = lane >> 4;
  const int qbase = blockIdx.x * 128 + wave * 16;

  __shared__ u16 Ks[32][72];  // [key][d], padded rows (144 B, 16B aligned)
  __shared__ u16 VsT[64][48]; // [d][key], padded rows (96 B, 16B aligned)

  // Q as B-fragments (B[k][n] = Q[query n][d k]); resident for whole pass.
  Frag qb[2];
#pragma unroll
  for (int c = 0; c < 2; ++c) {
    const u16* qr = Q + (size_t)(qbase + ln15) * NHD + c * 32 + half * 8;
    qb[c].h[0] = *(const u16x8*)qr;
    qb[c].h[1] = *(const u16x8*)(qr + 16);
  }

  f32x8 o[4] = {};
  float m_cur = -3.0e38f, l_cur = 0.0f; // stats for query qbase+ln15

  for (int kc = 0; kc < NS; kc += 32) {
    __syncthreads();
    { // K: straight copy via async DMA; V: VGPR path (transpose scatter)
      int r = tid >> 3, off = (tid & 7) * 8;
      async_copy_b128(&Ks[r][off], &K[(size_t)(kc + r) * NHD + off]);
      u16x8 vv = *(const u16x8*)&V[(size_t)(kc + r) * NHD + off];
#pragma unroll
      for (int e = 0; e < 8; ++e) VsT[off + e][r] = vv[e];
      if (kc + 32 < NS) {
        __builtin_prefetch(&K[(size_t)(kc + 32 + r) * NHD + off], 0, 3);
        __builtin_prefetch(&V[(size_t)(kc + 32 + r) * NHD + off], 0, 3);
      }
    }
    async_wait0();
    __syncthreads();

    // T = K_chunk @ Q^T : two 16x16 tiles (keys 0-15, 16-31), K-dim = 2x32
    f32x8 sT0 = {}, sT1 = {};
#pragma unroll
    for (int c = 0; c < 2; ++c) {
      Frag a0, a1; // A[row][k] = K[key row][d k] -> contiguous in d
      a0.h[0] = *(const u16x8*)&Ks[ln15][c * 32 + half * 8];
      a0.h[1] = *(const u16x8*)&Ks[ln15][c * 32 + 16 + half * 8];
      a1.h[0] = *(const u16x8*)&Ks[16 + ln15][c * 32 + half * 8];
      a1.h[1] = *(const u16x8*)&Ks[16 + ln15][c * 32 + 16 + half * 8];
      sT0 = bfwmma(a0, qb[c], sT0);
      sT1 = bfwmma(a1, qb[c], sT1);
    }
    // lane holds scores for query ln15, keys kc + j + 8*half (+16 for sT1)

    // online softmax per query (stats replicated across the half pair)
    float t = sT0[0];
#pragma unroll
    for (int j = 0; j < 8; ++j) {
      t = fmaxf(t, sT0[j]);
      t = fmaxf(t, sT1[j]);
    }
    t = fmaxf(t, __shfl_xor(t, 16, 32));
    float m_new = fmaxf(m_cur, t);

    Frag pf; // exp'd scores ARE the P A-fragment (in-lane repack)
    float p0[8], p1[8], ls = 0.0f;
#pragma unroll
    for (int j = 0; j < 8; ++j) {
      p0[j] = __expf(sT0[j] - m_new);
      p1[j] = __expf(sT1[j] - m_new);
      ls += p0[j] + p1[j];
    }
#pragma unroll
    for (int j = 0; j < 4; ++j) {
      pf.w[j] = f2bf_pk(p0[2 * j], p0[2 * j + 1]);
      pf.w[4 + j] = f2bf_pk(p1[2 * j], p1[2 * j + 1]);
    }
    ls += __shfl_xor(ls, 16, 32);
    float alpha = __expf(m_cur - m_new);
    l_cur = l_cur * alpha + ls;
    m_cur = m_new;

    // rescale O: row j of O is query j+8*half -> fetch that query's alpha
#pragma unroll
    for (int j = 0; j < 8; ++j) {
      float aj = __shfl(alpha, j + 8 * half, 32);
#pragma unroll
      for (int tn = 0; tn < 4; ++tn) o[tn][j] = o[tn][j] * aj;
    }

    // O += P @ V over HD = 4 n-tiles; B[k][n] = V[key k][d n] = VsT[d][k]
#pragma unroll
    for (int tn = 0; tn < 4; ++tn) {
      Frag vb;
      vb.h[0] = *(const u16x8*)&VsT[tn * 16 + ln15][half * 8];
      vb.h[1] = *(const u16x8*)&VsT[tn * 16 + ln15][16 + half * 8];
      o[tn] = bfwmma(pf, vb, o[tn]);
    }
  }

  float inv = 1.0f / l_cur;
#pragma unroll
  for (int j = 0; j < 8; ++j) {
    float ij = __shfl(inv, j + 8 * half, 32);
    int ss = qbase + j + 8 * half;
#pragma unroll
    for (int tn = 0; tn < 4; ++tn) {
      int dd = tn * 16 + ln15;
      Att[((size_t)bb * NS + ss) * NE + hh * NHD + dd] = f2bf(o[tn][j] * ij);
    }
  }
}

// ------------------------------------------------------- output projection
// P = Att(bf16) @ Wo + bo, Wo given transposed; f32 output for LayerNorm.
__global__ __launch_bounds__(256)
void oproj_kernel(const u16* __restrict__ A, const u16* __restrict__ Wt,
                  const float* __restrict__ bias, float* __restrict__ P) {
  __shared__ u16 As[128][32];
  __shared__ u16 BsT[64][32];

  const int tid = threadIdx.x;
  const int lane = tid & 31, wave = tid >> 5;
  const int wm = wave >> 1, wn = wave & 1;
  const int ln15 = lane & 15, half = lane >> 4;
  const int blockM = blockIdx.x * 128;
  const int blockN = blockIdx.y * 64;

  f32x8 acc[2][2] = {};

  for (int k0 = 0; k0 < NE; k0 += 32) {
    __syncthreads();
#pragma unroll
    for (int p = 0; p < 2; ++p) {
      int c = tid + p * 256;
      int r = c >> 2, off = (c & 3) * 8;
      async_copy_b128(&As[r][off], &A[(size_t)(blockM + r) * NE + k0 + off]);
    }
    {
      int r = tid >> 2, off = (tid & 3) * 8;
      async_copy_b128(&BsT[r][off], &Wt[(size_t)(blockN + r) * NE + k0 + off]);
    }
    async_wait0();
    __syncthreads();

    Frag a[2], b[2];
#pragma unroll
    for (int t = 0; t < 2; ++t) {
      int rA = wm * 32 + t * 16 + ln15;
      a[t].h[0] = *(const u16x8*)&As[rA][half * 8];
      a[t].h[1] = *(const u16x8*)&As[rA][16 + half * 8];
      int cB = wn * 32 + t * 16 + ln15;
      b[t].h[0] = *(const u16x8*)&BsT[cB][half * 8];
      b[t].h[1] = *(const u16x8*)&BsT[cB][16 + half * 8];
    }
#pragma unroll
    for (int tm = 0; tm < 2; ++tm)
#pragma unroll
      for (int tn = 0; tn < 2; ++tn)
        acc[tm][tn] = bfwmma(a[tm], b[tn], acc[tm][tn]);
  }

#pragma unroll
  for (int tm = 0; tm < 2; ++tm) {
#pragma unroll
    for (int tn = 0; tn < 2; ++tn) {
      int gn = blockN + wn * 32 + tn * 16 + ln15;
      float bias_v = bias[gn];
#pragma unroll
      for (int j = 0; j < 8; ++j) {
        int gm = blockM + wm * 32 + tm * 16 + j + 8 * half;
        P[(size_t)gm * NE + gn] = acc[tm][tn][j] + bias_v;
      }
    }
  }
}

// ----------------------------------------------------- residual + LayerNorm
__global__ __launch_bounds__(256)
void ln_kernel(const float* __restrict__ P, const float* __restrict__ X,
               const float* __restrict__ gamma, const float* __restrict__ beta,
               float* __restrict__ Out) {
  int wave = threadIdx.x >> 5, lane = threadIdx.x & 31;
  int rowid = blockIdx.x * 8 + wave;
  const float* pr = P + (size_t)rowid * NE;
  const float* xr = X + (size_t)rowid * NE;
  float v[16];
  float s = 0.0f;
#pragma unroll
  for (int i = 0; i < 16; ++i) {
    v[i] = pr[lane + i * 32] + xr[lane + i * 32];
    s += v[i];
  }
#pragma unroll
  for (int m = 1; m <= 16; m <<= 1) s += __shfl_xor(s, m, 32);
  float mean = s * (1.0f / NE);
  float var = 0.0f;
#pragma unroll
  for (int i = 0; i < 16; ++i) { float d = v[i] - mean; var += d * d; }
#pragma unroll
  for (int m = 1; m <= 16; m <<= 1) var += __shfl_xor(var, m, 32);
  var *= (1.0f / NE);
  float inv = rsqrtf(var + 1e-5f);
  float* orow = Out + (size_t)rowid * NE;
#pragma unroll
  for (int i = 0; i < 16; ++i) {
    int c = lane + i * 32;
    orow[c] = (v[i] - mean) * inv * gamma[c] + beta[c];
  }
}

// ------------------------------------------------------------------ launch
extern "C" void kernel_launch(void* const* d_in, const int* in_sizes, int n_in,
                              void* d_out, int out_size, void* d_ws,
                              size_t ws_size, hipStream_t stream) {
  (void)in_sizes; (void)n_in; (void)out_size; (void)ws_size;

  const float* xq = (const float*)d_in[0];
  const float* xk = (const float*)d_in[1];
  const float* xv = (const float*)d_in[2];
  const float* Wq = (const float*)d_in[3];
  const float* bq = (const float*)d_in[4];
  const float* Wk = (const float*)d_in[5];
  const float* bk = (const float*)d_in[6];
  const float* Wv = (const float*)d_in[7];
  const float* bv = (const float*)d_in[8];
  const float* Wo = (const float*)d_in[9];
  const float* bo = (const float*)d_in[10];
  const float* compat = (const float*)d_in[11];
  const float* gamma = (const float*)d_in[12];
  const float* beta = (const float*)d_in[13];

  char* ws = (char*)d_ws;
  const size_t SZ_X = (size_t)NM * NE * 2; // 8 MiB bf16 activation
  const size_t SZ_W = (size_t)NE * NE * 2; // 512 KiB bf16 weight
  u16* XQb = (u16*)(ws);
  u16* XKb = (u16*)(ws + SZ_X);
  u16* XVb = (u16*)(ws + 2 * SZ_X);
  u16* WQt = (u16*)(ws + 3 * SZ_X);
  u16* WKt = (u16*)(ws + 3 * SZ_X + SZ_W);
  u16* WVt = (u16*)(ws + 3 * SZ_X + 2 * SZ_W);
  u16* WOt = (u16*)(ws + 3 * SZ_X + 3 * SZ_W);
  u16* Qp = (u16*)(ws + 3 * SZ_X + 4 * SZ_W);
  u16* Kp = (u16*)(ws + 4 * SZ_X + 4 * SZ_W);
  u16* Vp = (u16*)(ws + 5 * SZ_X + 4 * SZ_W); // end = 50 MiB
  u16* Att = XQb;                    // alias: Xq_bf dead after QKV GEMM
  float* Proj = (float*)(ws + SZ_X); // alias: spans XKb+XVb (both dead)

  const int nAct = NM * NE;
  cvt_kernel<<<nAct / 1024, 256, 0, stream>>>(xq, XQb, nAct);
  cvt_kernel<<<nAct / 1024, 256, 0, stream>>>(xk, XKb, nAct);
  cvt_kernel<<<nAct / 1024, 256, 0, stream>>>(xv, XVb, nAct);
  dim3 tg(16, 16);
  cvt_t_kernel<<<tg, 256, 0, stream>>>(Wq, WQt);
  cvt_t_kernel<<<tg, 256, 0, stream>>>(Wk, WKt);
  cvt_t_kernel<<<tg, 256, 0, stream>>>(Wv, WVt);
  cvt_t_kernel<<<tg, 256, 0, stream>>>(Wo, WOt);

  qkv_gemm_kernel<<<dim3(NM / 128, NE / 64, 3), 256, 0, stream>>>(
      XQb, XKb, XVb, WQt, WKt, WVt, bq, bk, bv, compat, Qp, Kp, Vp);

  attn_kernel<<<dim3(NS / 128, NB * NH), 256, 0, stream>>>(Qp, Kp, Vp, Att);

  oproj_kernel<<<dim3(NM / 128, NE / 64), 256, 0, stream>>>(Att, WOt, bo, Proj);

  ln_kernel<<<NM / 8, 256, 0, stream>>>(Proj, xq, gamma, beta, (float*)d_out);
}
